// DeformableAlignment_18897856103182
// MI455X (gfx1250) — matrix operand order repaired and checked by reading, not compiled, not run on hardware
//
#include <hip/hip_runtime.h>
#include <hip/hip_bf16.h>

#define HH 128
#define WW 128
#define CH 64
#define BB 4
#define HWSZ (HH * WW)            // 16384
#define NPIX (BB * HWSZ)          // 65536
#define LDA 40                    // padded LDS row stride (bf16 elems), 80B

typedef __attribute__((ext_vector_type(16))) __bf16 bf16x16;
typedef __attribute__((ext_vector_type(8)))  __bf16 bf16x8;
typedef __attribute__((ext_vector_type(8)))  float  f32x8;

union Frag   { bf16x16 v; bf16x8 h[2]; };
union Pack16 { __bf16 b[16]; uint4 q[2]; };   // 16 bf16 packed for 2x ds_store_b128

// ---------------------------------------------------------------------------
// K0a: zero the per-channel stat accumulators (2 * 64 floats)
// ---------------------------------------------------------------------------
__global__ void init_stats_kernel(float* __restrict__ p) {
    p[threadIdx.x] = 0.0f;
}

// ---------------------------------------------------------------------------
// K0b/K0c: pre-pack conv weights (bf16) into the exact per-chunk LDS tile
// layout [chunk][n][kk].  144 KB total -> stays L2-resident; hot-loop weight
// staging becomes coalesced b128 copies instead of stride-9 scalar loads.
// ---------------------------------------------------------------------------
__global__ __launch_bounds__(256)
void pack_ow_kernel(const float* __restrict__ ow, __bf16* __restrict__ pw1) {
    const int idx = blockIdx.x * 256 + threadIdx.x;   // 36 chunks * 32n * 32k
    const int ci  = idx >> 10;
    const int n   = (idx >> 5) & 31;
    const int kk  = idx & 31;
    const int tap = ci >> 2;
    const int c0  = (ci & 3) * 32;
    const float v = (n < 27) ? ow[(n * 128 + c0 + kk) * 9 + tap] : 0.0f;
    pw1[idx] = (__bf16)v;
}

__global__ __launch_bounds__(256)
void pack_mw_kernel(const float* __restrict__ mw, __bf16* __restrict__ pw2) {
    const int idx = blockIdx.x * 256 + threadIdx.x;   // 18 chunks * 64n * 32k
    const int ci  = idx >> 11;
    const int n   = (idx >> 5) & 63;
    const int kk  = idx & 31;
    const int tap = ci >> 1;
    const int h   = ci & 1;
    pw2[idx] = (__bf16)mw[(n * CH + h * 32 + kk) * 9 + tap];
}

// ---------------------------------------------------------------------------
// K1: offset/mask conv3x3 over concat(f1,f3) as implicit GEMM with bf16 WMMA.
//   M = 64 pixels / WG (16 per wave), N = 32 (27 valid), K = 9 taps * 128 ch,
//   K-chunk = 32 (one tap, 32 channels). Branchless staging, packed LDS I/O.
// ---------------------------------------------------------------------------
__global__ __launch_bounds__(128)
void offset_conv_kernel(const float* __restrict__ f1, const float* __restrict__ f3,
                        const __bf16* __restrict__ pw1, const float* __restrict__ ob,
                        float* __restrict__ om) {
    __shared__ __bf16 a_lds[64 * LDA];
    __shared__ __bf16 b_lds[32 * LDA];

    const int tid  = threadIdx.x;
    const int wave = tid >> 5;
    const int lane = tid & 31;
    const int m    = lane & 15;
    const int hf   = lane >> 4;

    const int pixBase = blockIdx.x * 64;          // 64 consecutive px, same image row
    // A-staging map: thread -> (pixel, 16-channel slice)
    const int sp = tid >> 1;
    const int sc = (tid & 1) * 16;
    const int pix_s = pixBase + sp;
    const int b_s = pix_s >> 14;
    const int y_s = (pix_s >> 7) & 127;
    const int x_s = pix_s & 127;
    // B-staging map: thread -> 8 contiguous packed elems (= row tid>>2, cols (tid&3)*8)
    const int bn = tid >> 2;
    const int bk = (tid & 3) * 8;

    f32x8 acc0 = {}, acc1 = {};

    for (int tap = 0; tap < 9; ++tap) {
        const int yy = y_s + tap / 3 - 1, xx = x_s + tap % 3 - 1;
        const float inbf = ((yy >= 0) && (yy < HH) && (xx >= 0) && (xx < WW)) ? 1.0f : 0.0f;
        const int pclamp = min(max(yy, 0), HH - 1) * WW + min(max(xx, 0), WW - 1);

        for (int c0 = 0; c0 < 128; c0 += 32) {
            __syncthreads();  // previous chunk's fragment reads done
            // ---- stage A: slice is entirely in f1 or f3; unconditional clamped loads ----
            const int cbase = c0 + sc;
            const float* basep = (cbase < CH)
                ? f1 + (b_s * CH + cbase) * HWSZ + pclamp
                : f3 + (b_s * CH + (cbase - CH)) * HWSZ + pclamp;
            Pack16 pa;
            #pragma unroll
            for (int i = 0; i < 16; ++i)
                pa.b[i] = (__bf16)(basep[i * HWSZ] * inbf);
            uint4* adst = (uint4*)&a_lds[sp * LDA + sc];
            adst[0] = pa.q[0]; adst[1] = pa.q[1];
            // ---- stage B: coalesced copy of pre-packed bf16 weight tile ----
            const int ci = tap * 4 + (c0 >> 5);
            *(uint4*)&b_lds[bn * LDA + bk] =
                *(const uint4*)(pw1 + ci * 1024 + tid * 8);
            __syncthreads();
            // ---- fragments per documented CDNA5 16-bit layouts ----
            Frag a, b0, b1;
            const bf16x8* Arow = (const bf16x8*)&a_lds[(wave * 16 + m) * LDA];
            a.h[0] = Arow[hf];          // K = hf*8 .. hf*8+7
            a.h[1] = Arow[2 + hf];      // K = 16+hf*8 ..
            const bf16x8* B0 = (const bf16x8*)&b_lds[m * LDA];
            const bf16x8* B1 = (const bf16x8*)&b_lds[(16 + m) * LDA];
            b0.h[0] = B0[hf]; b0.h[1] = B0[2 + hf];   // K = hf*8 .. +7, 16+hf*8 .. +7
            b1.h[0] = B1[hf]; b1.h[1] = B1[2 + hf];
            acc0 = __builtin_amdgcn_wmma_f32_16x16x32_bf16(false, a.v, false, b0.v,
                                                           (short)0, acc0, false, false);
            acc1 = __builtin_amdgcn_wmma_f32_16x16x32_bf16(false, a.v, false, b1.v,
                                                           (short)0, acc1, false, false);
        }
    }

    // ---- epilogue: bias, sigmoid on mask channels, vectorized stores ----
    const int pix0 = pixBase + wave * 16;
    const int b_  = pix0 >> 14;
    const int rem = pix0 & 16383;   // multiple of 16 -> 8*hf offset is 32B aligned
    {   // N-tile 0: channels 0..15 (all offsets)
        const float bias = ob[m];
        float* dst = om + (b_ * 27 + m) * HWSZ + rem + 8 * hf;
        float4 lo, hi;
        lo.x = acc0[0] + bias; lo.y = acc0[1] + bias;
        lo.z = acc0[2] + bias; lo.w = acc0[3] + bias;
        hi.x = acc0[4] + bias; hi.y = acc0[5] + bias;
        hi.z = acc0[6] + bias; hi.w = acc0[7] + bias;
        *(float4*)dst = lo; *(float4*)(dst + 4) = hi;
    }
    {   // N-tile 1: channels 16..31 (16,17 offsets; 18..26 sigmoid mask; >=27 dropped)
        const int ch = 16 + m;
        if (ch < 27) {
            const float bias = ob[ch];
            const bool is_mask = ch >= 18;
            float v[8];
            #pragma unroll
            for (int r = 0; r < 8; ++r) {
                float t = acc1[r] + bias;
                v[r] = is_mask ? (1.0f / (1.0f + __expf(-t))) : t;
            }
            float* dst = om + (b_ * 27 + ch) * HWSZ + rem + 8 * hf;
            float4 lo = {v[0], v[1], v[2], v[3]};
            float4 hi = {v[4], v[5], v[6], v[7]};
            *(float4*)dst = lo; *(float4*)(dst + 4) = hi;
        }
    }
}

// ---------------------------------------------------------------------------
// K2: deformable bilinear sampling (L2-resident gathers) + main conv as
//     bf16 WMMA GEMM: M = 64 px / WG, N = 64 out-ch, K = 9 taps * 64 ch.
//     Also accumulates per-channel sum / sum-of-squares for the norm.
// ---------------------------------------------------------------------------
__global__ __launch_bounds__(128)
void deform_gemm_kernel(const float* __restrict__ f1, const float* __restrict__ om,
                        const __bf16* __restrict__ pw2, float* __restrict__ aligned,
                        float* __restrict__ chSum, float* __restrict__ chSq) {
    __shared__ __bf16 a_lds[64 * LDA];
    __shared__ __bf16 b_lds[64 * LDA];
    __shared__ float redS[64];
    __shared__ float redQ[64];

    const int tid  = threadIdx.x;
    const int wave = tid >> 5;
    const int lane = tid & 31;
    const int m    = lane & 15;
    const int hf   = lane >> 4;

    const int pixBase = blockIdx.x * 64;
    const int sp = tid >> 1;
    const int sc = (tid & 1) * 16;
    const int pix_s = pixBase + sp;
    const int b_s = pix_s >> 14;
    const int y_s = (pix_s >> 7) & 127;
    const int x_s = pix_s & 127;
    const float* f1b = f1 + b_s * CH * HWSZ;
    const float* omb = om + b_s * 27 * HWSZ;
    const int pp = y_s * WW + x_s;
    // B-staging map: thread -> 16 contiguous packed elems (= row tid>>1, cols (tid&1)*16)
    const int bn = tid >> 1;
    const int bk = (tid & 1) * 16;

    if (tid < 64) { redS[tid] = 0.0f; redQ[tid] = 0.0f; }

    f32x8 acc[4] = {{}, {}, {}, {}};

    for (int tap = 0; tap < 9; ++tap) {
        // ---- bilinear parameters for (pixel, tap): shared across all 64 channels ----
        const float offy = omb[(2 * tap)     * HWSZ + pp];
        const float offx = omb[(2 * tap + 1) * HWSZ + pp];
        const float msk  = omb[(18 + tap)    * HWSZ + pp];
        const float py = (float)(y_s + tap / 3 - 1) + offy;
        const float px = (float)(x_s + tap % 3 - 1) + offx;
        const float fy = floorf(py), fx = floorf(px);
        const int y0 = (int)fy, x0 = (int)fx;
        const float wy = py - fy, wx = px - fx;
        const int y0c = min(max(y0, 0), HH - 1),     y1c = min(max(y0 + 1, 0), HH - 1);
        const int x0c = min(max(x0, 0), WW - 1),     x1c = min(max(x0 + 1, 0), WW - 1);
        const float vy0 = (y0 >= 0 && y0 < HH) ? 1.0f : 0.0f;
        const float vy1 = (y0 + 1 >= 0 && y0 + 1 < HH) ? 1.0f : 0.0f;
        const float vx0 = (x0 >= 0 && x0 < WW) ? 1.0f : 0.0f;
        const float vx1 = (x0 + 1 >= 0 && x0 + 1 < WW) ? 1.0f : 0.0f;
        const float w00 = (1.0f - wy) * (1.0f - wx) * vy0 * vx0 * msk;
        const float w01 = (1.0f - wy) * wx          * vy0 * vx1 * msk;
        const float w10 = wy * (1.0f - wx)          * vy1 * vx0 * msk;
        const float w11 = wy * wx                   * vy1 * vx1 * msk;
        const int o00 = y0c * WW + x0c, o01 = y0c * WW + x1c;
        const int o10 = y1c * WW + x0c, o11 = y1c * WW + x1c;

        for (int h = 0; h < 2; ++h) {          // K-chunk = (tap, 32-channel half)
            __syncthreads();
            // ---- stage A: gather addresses invariant over c -> 4 L2 streams ----
            const float* pc0 = f1b + (h * 32 + sc) * HWSZ;
            Pack16 pa;
            #pragma unroll
            for (int i = 0; i < 16; ++i) {
                const float* pc = pc0 + i * HWSZ;
                pa.b[i] = (__bf16)(pc[o00] * w00 + pc[o01] * w01 +
                                   pc[o10] * w10 + pc[o11] * w11);
            }
            uint4* adst = (uint4*)&a_lds[sp * LDA + sc];
            adst[0] = pa.q[0]; adst[1] = pa.q[1];
            // ---- stage B: coalesced copy of pre-packed bf16 weight tile ----
            const int ci = tap * 2 + h;
            const uint4* s4 = (const uint4*)(pw2 + ci * 2048 + tid * 16);
            uint4* bdst = (uint4*)&b_lds[bn * LDA + bk];
            bdst[0] = s4[0]; bdst[1] = s4[1];
            __syncthreads();
            Frag a;
            const bf16x8* Arow = (const bf16x8*)&a_lds[(wave * 16 + m) * LDA];
            a.h[0] = Arow[hf];
            a.h[1] = Arow[2 + hf];
            #pragma unroll
            for (int nt = 0; nt < 4; ++nt) {
                Frag b;
                const bf16x8* Brow = (const bf16x8*)&b_lds[(nt * 16 + m) * LDA];
                b.h[0] = Brow[hf]; b.h[1] = Brow[2 + hf];
                acc[nt] = __builtin_amdgcn_wmma_f32_16x16x32_bf16(false, a.v, false, b.v,
                                                                  (short)0, acc[nt], false, false);
            }
        }
    }

    // ---- epilogue: vectorized store of aligned + per-channel sum/sumsq ----
    const int pix0 = pixBase + wave * 16;
    const int b_  = pix0 >> 14;
    const int rem = pix0 & 16383;
    #pragma unroll
    for (int nt = 0; nt < 4; ++nt) {
        const int ch = nt * 16 + m;
        float* dst = aligned + (b_ * CH + ch) * HWSZ + rem + 8 * hf;
        float4 lo = {acc[nt][0], acc[nt][1], acc[nt][2], acc[nt][3]};
        float4 hi = {acc[nt][4], acc[nt][5], acc[nt][6], acc[nt][7]};
        *(float4*)dst = lo; *(float4*)(dst + 4) = hi;
        float s = 0.0f, q = 0.0f;
        #pragma unroll
        for (int r = 0; r < 8; ++r) {
            const float v = acc[nt][r];
            s += v;
            q += v * v;
        }
        atomicAdd(&redS[ch], s);   // ds_add_f32
        atomicAdd(&redQ[ch], q);
    }
    __syncthreads();
    if (tid < 64) {
        atomicAdd(&chSum[tid], redS[tid]);   // global_atomic_add_f32
        atomicAdd(&chSq[tid],  redQ[tid]);
    }
}

// ---------------------------------------------------------------------------
// K3: per-channel normalization + affine, float4-vectorized (HW % 4 == 0)
// ---------------------------------------------------------------------------
__global__ __launch_bounds__(256)
void norm_kernel(const float* __restrict__ aligned, const float* __restrict__ chSum,
                 const float* __restrict__ chSq, const float* __restrict__ gamma,
                 const float* __restrict__ beta, float* __restrict__ out) {
    const int e = (blockIdx.x * 256 + threadIdx.x) * 4;
    const int ch = (e >> 14) & 63;                 // (e / HWSZ) % 64
    const float inv_n = 1.0f / (float)NPIX;
    const float mean = chSum[ch] * inv_n;
    const float var  = chSq[ch] * inv_n - mean * mean;
    const float sc = rsqrtf(var + 1e-5f) * gamma[ch];
    const float sh = beta[ch] - mean * sc;
    const float4 v = *(const float4*)(aligned + e);
    float4 o;
    o.x = v.x * sc + sh; o.y = v.y * sc + sh;
    o.z = v.z * sc + sh; o.w = v.w * sc + sh;
    *(float4*)(out + e) = o;
}

// ---------------------------------------------------------------------------
extern "C" void kernel_launch(void* const* d_in, const int* in_sizes, int n_in,
                              void* d_out, int out_size, void* d_ws, size_t ws_size,
                              hipStream_t stream) {
    const float* f1    = (const float*)d_in[0];
    const float* f3    = (const float*)d_in[1];
    const float* ow    = (const float*)d_in[2];
    const float* ob    = (const float*)d_in[3];
    const float* mw    = (const float*)d_in[4];
    const float* gamma = (const float*)d_in[5];
    const float* beta  = (const float*)d_in[6];
    float* out = (float*)d_out;

    float*  om      = (float*)d_ws;                        // B*27*H*W  (7.1 MB)
    float*  aligned = om + (size_t)BB * 27 * HWSZ;         // B*64*H*W  (16.8 MB)
    float*  stats   = aligned + (size_t)BB * CH * HWSZ;    // 128 floats
    __bf16* pw1     = (__bf16*)(stats + 128);              // 36 * 1024 bf16 (72 KB)
    __bf16* pw2     = pw1 + 36 * 1024;                     // 18 * 2048 bf16 (72 KB)

    init_stats_kernel<<<1, 128, 0, stream>>>(stats);
    pack_ow_kernel<<<(36 * 1024) / 256, 256, 0, stream>>>(ow, pw1);
    pack_mw_kernel<<<(18 * 2048) / 256, 256, 0, stream>>>(mw, pw2);
    offset_conv_kernel<<<NPIX / 64, 128, 0, stream>>>(f1, f3, pw1, ob, om);
    deform_gemm_kernel<<<NPIX / 64, 128, 0, stream>>>(f1, om, pw2, aligned,
                                                      stats, stats + 64);
    norm_kernel<<<(NPIX * CH) / (256 * 4), 256, 0, stream>>>(aligned, stats, stats + 64,
                                                             gamma, beta, out);
}
